// BiasedCrossAttention_9225589752631
// MI455X (gfx1250) — compile-verified, matrix operand order checked by reading, not run on hardware
//
#include <hip/hip_runtime.h>
#include <hip/hip_bf16.h>

typedef __attribute__((ext_vector_type(16))) __bf16 v16bf;
typedef __attribute__((ext_vector_type(8)))  __bf16 v8bf;
typedef __attribute__((ext_vector_type(8)))  float  v8f;

static constexpr int Bb  = 2;
static constexpr int Lq  = 1024;
static constexpr int Lk  = 1024;
static constexpr int Dm  = 1024;
static constexpr int Hh  = 16;
static constexpr int DHd = 64;

// ---------------------------------------------------------------------------
// WMMA helper: D = A(16x32 bf16) x B(32x16 bf16) + C(16x16 f32)
// ---------------------------------------------------------------------------
__device__ inline v8f wmma_bf16(v16bf a, v16bf b, v8f c) {
  return __builtin_amdgcn_wmma_f32_16x16x32_bf16(false, a, false, b,
                                                 (short)0, c, false, false);
}

// ---------------------------------------------------------------------------
// Load a 16x32 bf16 fragment (A-operand layout; identical addressing works for
// the B operand when the "columns of B" are rows of a row-major source).
// ISA layout (16-bit A 16x32): lane L holds row (L&15); halves 0..7 = K at
// ((L>>4)*8)..+7, halves 8..15 = K at 16+((L>>4)*8)..+7  -> two 16B loads.
// ---------------------------------------------------------------------------
__device__ inline v16bf load_frag(const __bf16* __restrict__ src, int rbase,
                                  int ld, int kb) {
  const int lane = threadIdx.x & 31;
  const int r    = rbase + (lane & 15);
  const int kh   = (lane >> 4) * 8;
  const __bf16* p = src + (size_t)r * ld + kb + kh;
  v8bf lo = *reinterpret_cast<const v8bf*>(p);
  v8bf hi = *reinterpret_cast<const v8bf*>(p + 16);
  v16bf o;
#pragma unroll
  for (int i = 0; i < 8; ++i) { o[i] = lo[i]; o[i + 8] = hi[i]; }
  return o;
}

// ---------------------------------------------------------------------------
// f32 -> bf16 conversion (grid-stride)
// ---------------------------------------------------------------------------
__global__ void f32_to_bf16_kernel(const float* __restrict__ in,
                                   __bf16* __restrict__ out, int n) {
  int i      = blockIdx.x * blockDim.x + threadIdx.x;
  int stride = gridDim.x * blockDim.x;
  for (; i < n; i += stride) out[i] = (__bf16)in[i];
}

// ---------------------------------------------------------------------------
// Tensor Data Mover: issue a 2D tile load (nrows x 32 bf16) from a row-major
// [tensor_rows x ld] bf16 matrix into LDS, per the CDNA5 D# layout
// (CDNA5 ISA). One issue per wave; tracked by TENSORcnt.
// This toolchain exposes the 6-arg builtin:
//   (uint32x4 g0, int32x8 g1, int32x4 g2, int32x4 g3, int32x8 g4, i32 cpol)
// ---------------------------------------------------------------------------
#define HAS_TDM __has_builtin(__builtin_amdgcn_tensor_load_to_lds)

#if HAS_TDM
typedef unsigned int v4u __attribute__((ext_vector_type(4)));
typedef int          v8i __attribute__((ext_vector_type(8)));
typedef int          v4i __attribute__((ext_vector_type(4)));

__device__ inline void tdm_load_tile(const __bf16* gsrc, int row0, int col0,
                                     int ld, int tensor_rows, int nrows,
                                     unsigned int lds_off) {
  const unsigned long long ga =
      (unsigned long long)(uintptr_t)(gsrc + (size_t)row0 * ld + col0);
  v4u g0;
  g0[0] = 1u;                                   // count=1, user desc
  g0[1] = lds_off;                              // lds_addr [63:32]
  g0[2] = (unsigned int)ga;                     // global_addr [95:64]
  g0[3] = (unsigned int)((ga >> 32) & 0x1FFFFFFu) | (2u << 30);  // addr|type=2
  v8i g1;
  g1[0] = (int)(1u << 16);                      // data_size=1 (2 bytes/elem)
  g1[1] = (int)(((unsigned)ld & 0xFFFFu) << 16);          // tensor_dim0 lo
  g1[2] = (int)((((unsigned)ld >> 16) & 0xFFFFu) |
                (((unsigned)tensor_rows & 0xFFFFu) << 16)); // dim0 hi|dim1 lo
  g1[3] = (int)((((unsigned)tensor_rows >> 16) & 0xFFFFu) |
                (32u << 16));                   // dim1 hi | tile_dim0=32
  g1[4] = nrows;                                // tile_dim1
  g1[5] = ld;                                   // tensor_dim0_stride [31:0]
  g1[6] = 0;
  g1[7] = 0;
  const v4i z4 = {0, 0, 0, 0};
  const v8i z8 = {0, 0, 0, 0, 0, 0, 0, 0};
  __builtin_amdgcn_tensor_load_to_lds(g0, g1, z4, z4, z8, 0);
}
#endif

// ---------------------------------------------------------------------------
// GEMM: C[m][n] = (sum_k A[m][k]*W[n][k] + bias[n]) * scale
//   A: bf16 [M,K] row-major, W: bf16 [N,K] row-major (i.e. X @ W^T)
//   mode 0: f32 out [M,N]
//   mode 1: bf16 out head-split [B,H,L,DH]            (Q/K projections)
//   mode 2: bf16 out head-split transposed [B,H,DH,L] (V projection)
// Block: 256 thr = 8 waves computing a 128x64 tile. Per 32-wide K-step the
// A (128x32) and W (64x32) tiles are staged in LDS, double-buffered via the
// Tensor Data Mover (wave 0 issues, s_wait_tensorcnt gates, barriers publish).
// ---------------------------------------------------------------------------
__global__ __launch_bounds__(256)
void gemm_wmma_bf16(const __bf16* __restrict__ A, const __bf16* __restrict__ W,
                    const float* __restrict__ bias, void* __restrict__ outp,
                    int M, int N, int K, int mode, float scale) {
  __shared__ __align__(16) __bf16 ldsA[2][128 * 32];  // 2 x 8KB
  __shared__ __align__(16) __bf16 ldsW[2][64 * 32];   // 2 x 4KB

  const int lane  = threadIdx.x & 31;
  const int wave  = threadIdx.x >> 5;
  const int mblk  = blockIdx.x * 128;
  const int nbase = blockIdx.y * 64;

  v8f acc[4] = {};

#if HAS_TDM
  if (wave == 0) {
    tdm_load_tile(A, mblk, 0, K, M, 128, (unsigned int)(uintptr_t)&ldsA[0][0]);
    tdm_load_tile(W, nbase, 0, K, N, 64, (unsigned int)(uintptr_t)&ldsW[0][0]);
    if (K > 32) {
      tdm_load_tile(A, mblk, 32, K, M, 128,
                    (unsigned int)(uintptr_t)&ldsA[1][0]);
      tdm_load_tile(W, nbase, 32, K, N, 64,
                    (unsigned int)(uintptr_t)&ldsW[1][0]);
    }
  }
  int cur = 0;
  for (int kb = 0; kb < K; kb += 32) {
    if (wave == 0) {
      if (kb + 32 < K) __builtin_amdgcn_s_wait_tensorcnt(2);  // cur buf done
      else             __builtin_amdgcn_s_wait_tensorcnt(0);
    }
    __syncthreads();  // publish cur buffer to all waves
    v16bf a = load_frag(&ldsA[cur][0], wave * 16, 32, 0);
#pragma unroll
    for (int t = 0; t < 4; ++t) {
      v16bf b = load_frag(&ldsW[cur][0], t * 16, 32, 0);
      acc[t] = wmma_bf16(a, b, acc[t]);
    }
    __syncthreads();  // all waves done reading before refill
    if (wave == 0 && kb + 64 < K) {
      tdm_load_tile(A, mblk, kb + 64, K, M, 128,
                    (unsigned int)(uintptr_t)&ldsA[cur][0]);
      tdm_load_tile(W, nbase, kb + 64, K, N, 64,
                    (unsigned int)(uintptr_t)&ldsW[cur][0]);
    }
    cur ^= 1;
  }
#else
  // Fallback: cooperative vector copies into LDS (no TDM builtin available).
  for (int kb = 0; kb < K; kb += 32) {
    for (int c = threadIdx.x; c < 128 * 4; c += 256) {  // A: 128 rows x 4 chunks
      const int r = c >> 2, cc = (c & 3) * 8;
      *(v8bf*)(&ldsA[0][r * 32 + cc]) =
          *(const v8bf*)(A + (size_t)(mblk + r) * K + kb + cc);
    }
    for (int c = threadIdx.x; c < 64 * 4; c += 256) {   // W: 64 rows x 4 chunks
      const int r = c >> 2, cc = (c & 3) * 8;
      *(v8bf*)(&ldsW[0][r * 32 + cc]) =
          *(const v8bf*)(W + (size_t)(nbase + r) * K + kb + cc);
    }
    __syncthreads();
    v16bf a = load_frag(&ldsA[0][0], wave * 16, 32, 0);
#pragma unroll
    for (int t = 0; t < 4; ++t) {
      v16bf b = load_frag(&ldsW[0][0], t * 16, 32, 0);
      acc[t] = wmma_bf16(a, b, acc[t]);
    }
    __syncthreads();
  }
#endif

  // C layout: lanes 0-15 hold n=lane, rows r(vgpr)=0..7; lanes 16-31 rows 8..15
  const int halfsel = lane >> 4;
  const int ncol    = lane & 15;
  const int mbase   = mblk + wave * 16;
#pragma unroll
  for (int t = 0; t < 4; ++t) {
#pragma unroll
    for (int r = 0; r < 8; ++r) {
      const int m = mbase + r + 8 * halfsel;
      const int n = nbase + t * 16 + ncol;
      const float v = (acc[t][r] + bias[n]) * scale;
      if (mode == 0) {
        ((float*)outp)[(size_t)m * N + n] = v;
      } else {
        const int b = m / Lq, l = m % Lq;
        const int h = n / DHd, dh = n % DHd;
        const __bf16 bv = (__bf16)v;
        if (mode == 1)
          ((__bf16*)outp)[(((size_t)b * Hh + h) * Lq + l) * DHd + dh] = bv;
        else
          ((__bf16*)outp)[(((size_t)b * Hh + h) * DHd + dh) * Lk + l] = bv;
      }
    }
  }
}

// ---------------------------------------------------------------------------
// Flash-style attention. One wave per (b, h, 16-query tile); 8 waves/block.
//   qh: [B,H,Lq,DH] bf16 (pre-scaled by 1/sqrt(DH))
//   kh: [B,H,Lk,DH] bf16
//   vT: [B,H,DH,Lk] bf16   (transposed so B-operand K is contiguous)
//   lb: [B,Lq,Lk] f32 logits bias, kpm: [B,Lk] bool (byte) padding mask
//   out: [B,Lq,D] bf16 (heads merged) — feeds the Wo GEMM
// ---------------------------------------------------------------------------
__global__ __launch_bounds__(256)
void attn_wmma(const __bf16* __restrict__ qh, const __bf16* __restrict__ kh,
               const __bf16* __restrict__ vT, const float* __restrict__ lb,
               const unsigned char* __restrict__ kpm,
               __bf16* __restrict__ attn_out) {
  __shared__ __align__(16) __bf16 plds[8][16][32];  // P staging, 1KB/wave

  const int lane = threadIdx.x & 31;
  const int wave = threadIdx.x >> 5;
  const int w    = blockIdx.x * 8 + wave;       // 2048 waves total
  const int qtile = w & 63;                     // Lq/16 = 64
  const int h     = (w >> 6) & 15;
  const int b     = w >> 10;
  const int qbase = qtile * 16;

  const __bf16* qp = qh + (((size_t)b * Hh + h) * Lq) * DHd;
  const __bf16* kp = kh + (((size_t)b * Hh + h) * Lk) * DHd;
  const __bf16* vp = vT + (((size_t)b * Hh + h) * DHd) * Lk;
  const float*  lbp = lb + ((size_t)b * Lq + qbase) * Lk;
  const unsigned char* mp = kpm + (size_t)b * Lk;

  const v16bf q0 = load_frag(qp, qbase, DHd, 0);
  const v16bf q1 = load_frag(qp, qbase, DHd, 32);

  v8f oacc[4] = {};
  float mrow[8], lrow[8];
#pragma unroll
  for (int r = 0; r < 8; ++r) { mrow[r] = -1e30f; lrow[r] = 0.f; }

  const int halfsel = lane >> 4;
  const int ncol    = lane & 15;

  for (int kb = 0; kb < Lk; kb += 32) {
    // ---- S = q . k^T for 32 keys (two 16-key subtiles), K=DH=64 ----
    v16bf k0 = load_frag(kp, kb,      DHd, 0);
    v16bf k1 = load_frag(kp, kb,      DHd, 32);
    v16bf k2 = load_frag(kp, kb + 16, DHd, 0);
    v16bf k3 = load_frag(kp, kb + 16, DHd, 32);
    v8f s0 = {}, s1 = {};
    s0 = wmma_bf16(q0, k0, s0);
    s0 = wmma_bf16(q1, k1, s0);
    s1 = wmma_bf16(q0, k2, s1);
    s1 = wmma_bf16(q1, k3, s1);

    // ---- bias + key padding mask (attn_mask = mask ? -inf : bias) ----
    const int key0 = kb + ncol;
    const int key1 = kb + 16 + ncol;
    const bool msk0 = mp[key0] != 0;
    const bool msk1 = mp[key1] != 0;
#pragma unroll
    for (int r = 0; r < 8; ++r) {
      const int qrow = r + 8 * halfsel;
      s0[r] += msk0 ? -1e30f : lbp[(size_t)qrow * Lk + key0];
      s1[r] += msk1 ? -1e30f : lbp[(size_t)qrow * Lk + key1];
    }

    // ---- online softmax: row stats via width-16 shuffles ----
    float p0v[8], p1v[8];
#pragma unroll
    for (int r = 0; r < 8; ++r) {
      float v = fmaxf(s0[r], s1[r]);
#pragma unroll
      for (int off = 8; off >= 1; off >>= 1)
        v = fmaxf(v, __shfl_xor(v, off, 16));
      const float mnew = fmaxf(mrow[r], v);
      const float cf   = __expf(mrow[r] - mnew);
      mrow[r] = mnew;
      const float p0 = __expf(s0[r] - mnew);
      const float p1 = __expf(s1[r] - mnew);
      float rs = p0 + p1;
#pragma unroll
      for (int off = 8; off >= 1; off >>= 1)
        rs += __shfl_xor(rs, off, 16);
      lrow[r] = lrow[r] * cf + rs;
#pragma unroll
      for (int t = 0; t < 4; ++t) oacc[t][r] *= cf;
      p0v[r] = p0;
      p1v[r] = p1;
    }

    // ---- reshape P: C-layout -> A-fragment layout through LDS ----
#pragma unroll
    for (int r = 0; r < 8; ++r) {
      const int qrow = r + 8 * halfsel;
      plds[wave][qrow][ncol]      = (__bf16)p0v[r];
      plds[wave][qrow][16 + ncol] = (__bf16)p1v[r];
    }
    asm volatile("s_wait_dscnt 0" ::: "memory");  // wave-local LDS RAW
    const v16bf pf = load_frag(&plds[wave][0][0], 0, 32, 0);

    // ---- O += P(16x32) x V(32x64): 4 n-tiles, contiguous rows of vT ----
#pragma unroll
    for (int t = 0; t < 4; ++t) {
      v16bf vf = load_frag(vp, t * 16, Lk, kb);
      oacc[t] = wmma_bf16(pf, vf, oacc[t]);
    }
  }

  // ---- epilogue: normalize, merge heads, bf16 out [B,Lq,D] ----
#pragma unroll
  for (int t = 0; t < 4; ++t) {
#pragma unroll
    for (int r = 0; r < 8; ++r) {
      const int qrow = qbase + r + 8 * halfsel;
      const int dh   = t * 16 + ncol;
      const float v  = oacc[t][r] / lrow[r];
      attn_out[((size_t)b * Lq + qrow) * Dm + h * DHd + dh] = (__bf16)v;
    }
  }
}

// ---------------------------------------------------------------------------
// Host launch. Inputs (setup_inputs order):
//  0:q 1:k 2:v 3:Wq 4:bq 5:Wk 6:bk 7:Wv 8:bv 9:Wo 10:bo 11:logits_bias 12:kpm
// ---------------------------------------------------------------------------
extern "C" void kernel_launch(void* const* d_in, const int* in_sizes, int n_in,
                              void* d_out, int out_size, void* d_ws,
                              size_t ws_size, hipStream_t stream) {
  (void)in_sizes; (void)n_in; (void)out_size; (void)ws_size;

  const float* q  = (const float*)d_in[0];
  const float* k  = (const float*)d_in[1];
  const float* v  = (const float*)d_in[2];
  const float* Wq = (const float*)d_in[3];
  const float* bq = (const float*)d_in[4];
  const float* Wk = (const float*)d_in[5];
  const float* bk = (const float*)d_in[6];
  const float* Wv = (const float*)d_in[7];
  const float* bv = (const float*)d_in[8];
  const float* Wo = (const float*)d_in[9];
  const float* bo = (const float*)d_in[10];
  const float* lb = (const float*)d_in[11];
  const unsigned char* kpm = (const unsigned char*)d_in[12];

  char* ws = (char*)d_ws;
  const size_t MB = 1ull << 20;
  __bf16* xq  = (__bf16*)(ws + 0 * MB);   // [2048,1024] bf16  (4MB)
  __bf16* xk  = (__bf16*)(ws + 4 * MB);
  __bf16* xv  = (__bf16*)(ws + 8 * MB);
  __bf16* wqb = (__bf16*)(ws + 12 * MB);  // [1024,1024] bf16  (2MB each)
  __bf16* wkb = (__bf16*)(ws + 14 * MB);
  __bf16* wvb = (__bf16*)(ws + 16 * MB);
  __bf16* wob = (__bf16*)(ws + 18 * MB);
  __bf16* qhb = (__bf16*)(ws + 20 * MB);  // [B,H,Lq,DH] bf16  (4MB)
  __bf16* khb = (__bf16*)(ws + 24 * MB);  // [B,H,Lk,DH]
  __bf16* vtb = (__bf16*)(ws + 28 * MB);  // [B,H,DH,Lk]
  __bf16* aob = (__bf16*)(ws + 32 * MB);  // [B,Lq,D]

  const int nx = Bb * Lq * Dm;  // 2M elements
  const int nw = Dm * Dm;       // 1M elements
  f32_to_bf16_kernel<<<4096, 256, 0, stream>>>(q,  xq,  nx);
  f32_to_bf16_kernel<<<4096, 256, 0, stream>>>(k,  xk,  nx);
  f32_to_bf16_kernel<<<4096, 256, 0, stream>>>(v,  xv,  nx);
  f32_to_bf16_kernel<<<2048, 256, 0, stream>>>(Wq, wqb, nw);
  f32_to_bf16_kernel<<<2048, 256, 0, stream>>>(Wk, wkb, nw);
  f32_to_bf16_kernel<<<2048, 256, 0, stream>>>(Wv, wvb, nw);
  f32_to_bf16_kernel<<<2048, 256, 0, stream>>>(Wo, wob, nw);

  const int M = Bb * Lq;  // 2048
  const dim3 gg(M / 128, Dm / 64);  // (16,16)
  const float scale = 0.125f;       // 1/sqrt(DH), folded into Q projection
  gemm_wmma_bf16<<<gg, 256, 0, stream>>>(xq, wqb, bq, qhb, M, Dm, Dm, 1, scale);
  gemm_wmma_bf16<<<gg, 256, 0, stream>>>(xk, wkb, bk, khb, M, Dm, Dm, 1, 1.0f);
  gemm_wmma_bf16<<<gg, 256, 0, stream>>>(xv, wvb, bv, vtb, M, Dm, Dm, 2, 1.0f);

  attn_wmma<<<Bb * Hh * (Lq / 16) / 8, 256, 0, stream>>>(qhb, khb, vtb, lb,
                                                         kpm, aob);

  gemm_wmma_bf16<<<gg, 256, 0, stream>>>(aob, wob, bo, d_out, M, Dm, Dm, 0,
                                         1.0f);
}
